// SelfAttention_66082366816681
// MI455X (gfx1250) — compile-verified, matrix operand order checked by reading, not run on hardware
//
#include <hip/hip_runtime.h>
#include <hip/hip_bf16.h>

// ---------------------------------------------------------------------------
// Self-attention (B=8, N=1024, DIM=768, H=12, Dh=64) on gfx1250 (MI455X).
// - All matrix math via v_wmma_f32_16x16x32_bf16 (wave32).
// - Flash-style online softmax: 1024x1024 score matrices never touch HBM.
// - Attention K/V tiles staged into LDS via the Tensor Data Mover
//   (tensor_load_to_lds + s_wait_tensorcnt), double-buffered so the DMA of
//   the next tile overlaps the WMMA work on the current one.
// ---------------------------------------------------------------------------

typedef __attribute__((ext_vector_type(16))) __bf16        v16bf;
typedef __attribute__((ext_vector_type(8)))  float         v8f;
typedef __attribute__((ext_vector_type(4)))  unsigned int  v4u;
typedef __attribute__((ext_vector_type(8)))  int           v8i;
typedef __attribute__((ext_vector_type(4)))  int           v4i;

#define DIM    768
#define HEADS  12
#define DH     64
#define B_SZ   8
#define N_SEQ  1024
#define TOKENS (B_SZ * N_SEQ)
#define QK_SCALE 0.125f   // 64^-0.5

// f32 -> bf16 with round-to-nearest-even
static __device__ __forceinline__ unsigned short f2bf(float f) {
  union { float f; unsigned int u; } c; c.f = f;
  unsigned int u = c.u;
  u += 0x7FFFu + ((u >> 16) & 1u);
  return (unsigned short)(u >> 16);
}

union FragU { v16bf v; uint4 q[2]; };

// Load one 16x32 bf16 A-frag (or mirrored B-frag) from a row-major tile.
// ISA A-layout (16-bit, 16x32): lane l (l<16): M=l, K={0..7,16..23};
// lane l+16: M=l, K={8..15,24..31}. Two contiguous 16B chunks per lane.
static __device__ __forceinline__ v16bf load_frag16(const unsigned short* base, int ld) {
  const int lane = threadIdx.x;           // blocks are dim3(32, waves)
  const int row = lane & 15;
  const int g   = lane >> 4;
  const unsigned short* p = base + (size_t)row * ld + g * 8;
  FragU f;
  f.q[0] = *(const uint4*)(p);            // K = 8g .. 8g+7
  f.q[1] = *(const uint4*)(p + 16);       // K = 16+8g .. 16+8g+7
  return f.v;
}

static __device__ __forceinline__ v8f wmma_bf16(v16bf a, v16bf b, v8f c) {
  return __builtin_amdgcn_wmma_f32_16x16x32_bf16(
      /*neg_a=*/false, a, /*neg_b=*/false, b,
      /*c_mod=*/(short)0, c, /*reuse_a=*/false, /*reuse_b=*/false);
}

// LDS byte offset of a __shared__ object: in the flat LDS aperture the low
// 32 bits of the generic address carry the LDS offset (ISA 10.2 aperture map).
static __device__ __forceinline__ unsigned lds_off(const void* p) {
  return (unsigned)(size_t)p;
}

// ------------------------- Tensor Data Mover (TDM) -------------------------
// Build a D# for a 2-D tile of 16-bit elements and issue TENSOR_LOAD_TO_LDS.
// D# group0: count=1 | lds_addr | global_addr[56:0] | type=2.
// D# group1: data_size=2B, tensor_dim0/1, tile_dim0/1, tensor_dim0_stride.
// 1-D copies: pass tile_d1 = 0 (ISA: tile_dim1==0 -> unused).
static __device__ __forceinline__ void tdm_load_2d_b16(
    unsigned lds_addr, const void* gptr,
    unsigned tile_d0, unsigned tile_d1,
    unsigned tensor_d0, unsigned tensor_d1, unsigned d0_stride_elems) {
  const unsigned long long ga = (unsigned long long)(size_t)gptr;
  v4u g0;
  g0[0] = 1u;                                          // count=1, user mode
  g0[1] = lds_addr;                                    // LDS byte address
  g0[2] = (unsigned)(ga & 0xFFFFFFFFu);                // global_addr[31:0]
  g0[3] = (unsigned)((ga >> 32) & 0x01FFFFFFu)         // global_addr[56:32]
          | 0x80000000u;                               // type = 2 ("image")
  v8i g1;
  g1[0] = 0x00010000;                                  // data_size=1 (2 bytes)
  g1[1] = (int)((tensor_d0 & 0xFFFFu) << 16);          // tensor_dim0[15:0]
  g1[2] = (int)(((tensor_d0 >> 16) & 0xFFFFu) |
                ((tensor_d1 & 0xFFFFu) << 16));        // td0[31:16] | td1[15:0]
  g1[3] = (int)(((tensor_d1 >> 16) & 0xFFFFu) |
                ((tile_d0 & 0xFFFFu) << 16));          // td1[31:16] | tile_d0
  g1[4] = (int)(tile_d1 & 0xFFFFu);                    // tile_d1 | tile_d2=0
  g1[5] = (int)d0_stride_elems;                        // tensor_dim0_stride lo
  g1[6] = 0;                                           // stride hi | d1_stride lo
  g1[7] = 0;
  const v4i z4 = {0, 0, 0, 0};
#if __clang_major__ >= 23
  const v8i z8 = {0, 0, 0, 0, 0, 0, 0, 0};
  __builtin_amdgcn_tensor_load_to_lds(g0, g1, z4, z4, z8, 0);
#else
  __builtin_amdgcn_tensor_load_to_lds(g0, g1, z4, z4, 0);
#endif
}

// --------------------------- conversion kernels ----------------------------

__global__ void k_f32_to_bf16(const float* __restrict__ src,
                              unsigned short* __restrict__ dst, int n) {
  int i = blockIdx.x * blockDim.x + threadIdx.x;
  if (i < n) dst[i] = f2bf(src[i]);
}

// src: [in=768][out=768] f32 row-major  ->  dst: [out][in] bf16
__global__ void k_transpose_bf16(const float* __restrict__ src,
                                 unsigned short* __restrict__ dst) {
  int i = blockIdx.x * blockDim.x + threadIdx.x;
  if (i < DIM * DIM) {
    int o = i / DIM, in = i % DIM;
    dst[(size_t)o * DIM + in] = f2bf(src[(size_t)in * DIM + o]);
  }
}

// ------------------------------ QKV GEMM -----------------------------------
// D[m=token, n=outfeat] = sum_k x[m,k] * wT[n,k] + bias[n]
// vmode==0: store [B,H,N,Dh] row-major (Q and K)
// vmode==1: store transposed [B,H,Dh,N]  (V^T, feeds PV B-frags contiguously)
__global__ __launch_bounds__(128)
void k_qkv_gemm(const unsigned short* __restrict__ xbf,
                const unsigned short* __restrict__ wT,
                const float* __restrict__ bias,
                unsigned short* __restrict__ dst, int vmode) {
  const int lane = threadIdx.x;
  const int m0 = (blockIdx.x * 4 + threadIdx.y) * 16;  // token tile
  const int n0 = blockIdx.y * 64;                      // out-feature tile
  v8f acc[4] = {};
  for (int kk = 0; kk < DIM; kk += 32) {
    const v16bf a = load_frag16(xbf + (size_t)m0 * DIM + kk, DIM);
#pragma unroll
    for (int j = 0; j < 4; ++j) {
      const v16bf b = load_frag16(wT + (size_t)(n0 + j * 16) * DIM + kk, DIM);
      acc[j] = wmma_bf16(a, b, acc[j]);
    }
  }
  const int g = lane >> 4, nl = lane & 15;
#pragma unroll
  for (int j = 0; j < 4; ++j) {
    const int of = n0 + j * 16 + nl;      // out feature = h*64 + d
    const int h = of >> 6, d = of & 63;
    const float bv = bias[of];
    if (vmode) {
      // V^T store: fixed d per lane, 8 consecutive positions -> one b128
      const int tok0 = m0 + 8 * g;
      const int b = tok0 >> 10, pos0 = tok0 & 1023;
      union { uint4 u4; unsigned short us[8]; } pk;
#pragma unroll
      for (int r = 0; r < 8; ++r) pk.us[r] = f2bf(acc[j][r] + bv);
      *(uint4*)(dst + ((((size_t)b * HEADS + h) * DH + d) * N_SEQ + pos0)) = pk.u4;
    } else {
#pragma unroll
      for (int r = 0; r < 8; ++r) {
        const int tok = m0 + r + 8 * g;
        const int b = tok >> 10, pos = tok & 1023;
        dst[(((size_t)b * HEADS + h) * N_SEQ + pos) * DH + d] = f2bf(acc[j][r] + bv);
      }
    }
  }
}

// ---------------------------- flash attention ------------------------------
// Grid: (N/128, H, B); block dim3(32,8): each wave owns a 16-row Q tile.
// All 8 waves share each K/V tile, staged into LDS by the TDM (wave 0 issues
// the descriptors; double-buffered so DMA of tile i+1 overlaps compute on i).
__global__ __launch_bounds__(256)
void k_attention(const unsigned short* __restrict__ q,
                 const unsigned short* __restrict__ k,
                 const unsigned short* __restrict__ vT,
                 const unsigned char* __restrict__ mask,
                 unsigned short* __restrict__ attn_out) {
  const int b = blockIdx.z, h = blockIdx.y;
  const int lane = threadIdx.x, wv = threadIdx.y;
  const int m0 = blockIdx.x * 128 + wv * 16;

  __shared__ unsigned char  smask[N_SEQ];
  // [buf][ K tile 32x64 (2048 el) | V^T tile 64x32 (2048 el) ]
  __shared__ unsigned short skv[2][4096];
  __shared__ unsigned short sp[8][16 * 32];   // per-wave P staging (C->A relayout)

  for (int i = wv * 32 + lane; i < N_SEQ; i += 256)
    smask[i] = mask[(size_t)b * N_SEQ + i];

  const size_t bh = (size_t)b * HEADS + h;
  const unsigned short* qb = q  + bh * N_SEQ * DH;
  const unsigned short* kb = k  + bh * N_SEQ * DH;
  const unsigned short* vb = vT + bh * DH * N_SEQ;

  // Prologue: DMA tile 0 into buffer 0 (K is contiguous: 1-D 2048-element
  // copy; V^T is a 2-D tile: 64 rows of 32 elements, row stride 1024).
  if (wv == 0) {
    tdm_load_2d_b16(lds_off(&skv[0][0]),    kb, 2048, 0, 2048, 1, 2048);
    tdm_load_2d_b16(lds_off(&skv[0][2048]), vb, 32, 64, N_SEQ, DH, N_SEQ);
    __builtin_amdgcn_s_wait_tensorcnt(0);
  }
  __syncthreads();

  const v16bf qa0 = load_frag16(qb + (size_t)m0 * DH,      DH);  // feats 0..31
  const v16bf qa1 = load_frag16(qb + (size_t)m0 * DH + 32, DH);  // feats 32..63

  v8f o0 = {}, o1 = {}, o2 = {}, o3 = {};
  float mi[8], li[8];
#pragma unroll
  for (int r = 0; r < 8; ++r) { mi[r] = -__builtin_inff(); li[r] = 0.f; }

  const int g = lane >> 4, nl = lane & 15;
  unsigned short* pw = &sp[wv][0];

  for (int kv0 = 0, it = 0; kv0 < N_SEQ; kv0 += 32, ++it) {
    const int buf = it & 1;
    // Prefetch next kv tile into the other buffer (safe: the barrier at the
    // end of the previous iteration released all readers of buf^1).
    if (wv == 0 && kv0 + 32 < N_SEQ) {
      tdm_load_2d_b16(lds_off(&skv[buf ^ 1][0]),
                      kb + (size_t)(kv0 + 32) * DH, 2048, 0, 2048, 1, 2048);
      tdm_load_2d_b16(lds_off(&skv[buf ^ 1][2048]),
                      vb + (kv0 + 32), 32, 64, N_SEQ, DH, N_SEQ);
    }
    const unsigned short* kt = &skv[buf][0];      // [32][64] row-major
    const unsigned short* vt = &skv[buf][2048];   // [64][32] row-major (d, pos)

    // S = Q K^T for 32 kv positions (two 16-wide tiles, K-dim 64 = 2 wmma each)
    v8f s0 = {}, s1 = {};
    {
      const v16bf kb0l = load_frag16(kt + 0 * DH,       DH);
      const v16bf kb0h = load_frag16(kt + 0 * DH + 32,  DH);
      s0 = wmma_bf16(qa0, kb0l, s0);
      s0 = wmma_bf16(qa1, kb0h, s0);
      const v16bf kb1l = load_frag16(kt + 16 * DH,      DH);
      const v16bf kb1h = load_frag16(kt + 16 * DH + 32, DH);
      s1 = wmma_bf16(qa0, kb1l, s1);
      s1 = wmma_bf16(qa1, kb1h, s1);
    }

    const bool msk0 = smask[kv0 + nl]      != 0;  // per-column (per-lane) mask
    const bool msk1 = smask[kv0 + 16 + nl] != 0;

    // Online softmax update. C-layout: lane holds 8 rows (m = r + 8g), one
    // column each; row stats reduce across the 16-lane half via shfl_xor.
#pragma unroll
    for (int r = 0; r < 8; ++r) {
      float a0 = msk0 ? -10000.f : s0[r] * QK_SCALE;
      float a1 = msk1 ? -10000.f : s1[r] * QK_SCALE;
      float pm = fmaxf(a0, a1);
#pragma unroll
      for (int x = 8; x >= 1; x >>= 1) pm = fmaxf(pm, __shfl_xor(pm, x, 16));
      const float mnew = fmaxf(mi[r], pm);
      const float p0 = __expf(a0 - mnew);
      const float p1 = __expf(a1 - mnew);
      float rs = p0 + p1;
#pragma unroll
      for (int x = 8; x >= 1; x >>= 1) rs += __shfl_xor(rs, x, 16);
      const float corr = __expf(mi[r] - mnew);
      li[r] = li[r] * corr + rs;
      mi[r] = mnew;
      o0[r] *= corr; o1[r] *= corr; o2[r] *= corr; o3[r] *= corr;
      const int row = r + 8 * g;
      pw[row * 32 + nl]      = f2bf(p0);   // P in row-major [16 x 32] (LDS)
      pw[row * 32 + 16 + nl] = f2bf(p1);
    }

    // O += P(16x32) * V(32x16) per 16-wide d-slice; P re-loaded in A-layout
    const v16bf pa  = load_frag16(pw, 32);               // ds_load_b128 x2
    const v16bf vb0 = load_frag16(vt + 0 * 16 * 32, 32);
    o0 = wmma_bf16(pa, vb0, o0);
    const v16bf vb1 = load_frag16(vt + 1 * 16 * 32, 32);
    o1 = wmma_bf16(pa, vb1, o1);
    const v16bf vb2 = load_frag16(vt + 2 * 16 * 32, 32);
    o2 = wmma_bf16(pa, vb2, o2);
    const v16bf vb3 = load_frag16(vt + 3 * 16 * 32, 32);
    o3 = wmma_bf16(pa, vb3, o3);

    // Buffer handoff: producer drains TENSORcnt, then whole block syncs.
    if (wv == 0) __builtin_amdgcn_s_wait_tensorcnt(0);
    __syncthreads();
  }

  // Normalize and store [B, N, H*Dh] bf16 for the output projection.
#pragma unroll
  for (int r = 0; r < 8; ++r) {
    const float inv = 1.f / li[r];
    const int tok = b * N_SEQ + m0 + r + 8 * g;
    unsigned short* orow = attn_out + (size_t)tok * DIM + h * DH;
    orow[0  + nl] = f2bf(o0[r] * inv);
    orow[16 + nl] = f2bf(o1[r] * inv);
    orow[32 + nl] = f2bf(o2[r] * inv);
    orow[48 + nl] = f2bf(o3[r] * inv);
  }
}

// --------------------------- output projection -----------------------------
__global__ __launch_bounds__(128)
void k_out_proj(const unsigned short* __restrict__ abf,
                const unsigned short* __restrict__ woT,
                const float* __restrict__ bo,
                float* __restrict__ out) {
  const int lane = threadIdx.x;
  const int m0 = (blockIdx.x * 4 + threadIdx.y) * 16;
  const int n0 = blockIdx.y * 64;
  v8f acc[4] = {};
  for (int kk = 0; kk < DIM; kk += 32) {
    const v16bf a = load_frag16(abf + (size_t)m0 * DIM + kk, DIM);
#pragma unroll
    for (int j = 0; j < 4; ++j) {
      const v16bf b = load_frag16(woT + (size_t)(n0 + j * 16) * DIM + kk, DIM);
      acc[j] = wmma_bf16(a, b, acc[j]);
    }
  }
  const int g = lane >> 4, nl = lane & 15;
#pragma unroll
  for (int j = 0; j < 4; ++j) {
    const int of = n0 + j * 16 + nl;
    const float bv = bo[of];
#pragma unroll
    for (int r = 0; r < 8; ++r) {
      const int tok = m0 + r + 8 * g;
      out[(size_t)tok * DIM + of] = acc[j][r] + bv;
    }
  }
}

// ------------------------------- launcher ----------------------------------

extern "C" void kernel_launch(void* const* d_in, const int* in_sizes, int n_in,
                              void* d_out, int out_size, void* d_ws, size_t ws_size,
                              hipStream_t stream) {
  (void)in_sizes; (void)n_in; (void)out_size; (void)ws_size;
  const float*         x    = (const float*)d_in[0];
  const unsigned char* mask = (const unsigned char*)d_in[1];  // jnp bool = 1 byte
  const float* wq = (const float*)d_in[2];
  const float* bq = (const float*)d_in[3];
  const float* wk = (const float*)d_in[4];
  const float* bk = (const float*)d_in[5];
  const float* wv = (const float*)d_in[6];
  const float* bv = (const float*)d_in[7];
  const float* wo = (const float*)d_in[8];
  const float* bo = (const float*)d_in[9];
  float* out = (float*)d_out;

  // workspace carve-up (bf16 intermediates): ~64.5 MB total
  char* ws = (char*)d_ws;
  const size_t XBF = (size_t)TOKENS * DIM * 2;                 // 12.6 MB
  const size_t WT  = (size_t)DIM * DIM * 2;                    // 1.18 MB
  const size_t QKV = (size_t)B_SZ * HEADS * N_SEQ * DH * 2;    // 12.6 MB
  unsigned short* xbf  = (unsigned short*)ws; ws += XBF;
  unsigned short* wqT  = (unsigned short*)ws; ws += WT;
  unsigned short* wkT  = (unsigned short*)ws; ws += WT;
  unsigned short* wvT  = (unsigned short*)ws; ws += WT;
  unsigned short* woT  = (unsigned short*)ws; ws += WT;
  unsigned short* qbuf = (unsigned short*)ws; ws += QKV;       // [B,H,N,Dh]
  unsigned short* kbuf = (unsigned short*)ws; ws += QKV;       // [B,H,N,Dh]
  unsigned short* vTbf = (unsigned short*)ws; ws += QKV;       // [B,H,Dh,N]
  unsigned short* aout = (unsigned short*)ws; ws += XBF;       // [B,N,H*Dh]

  k_f32_to_bf16<<<(TOKENS * DIM + 255) / 256, 256, 0, stream>>>(x, xbf, TOKENS * DIM);
  const int tg = (DIM * DIM + 255) / 256;
  k_transpose_bf16<<<tg, 256, 0, stream>>>(wq, wqT);
  k_transpose_bf16<<<tg, 256, 0, stream>>>(wk, wkT);
  k_transpose_bf16<<<tg, 256, 0, stream>>>(wv, wvT);
  k_transpose_bf16<<<tg, 256, 0, stream>>>(wo, woT);

  const dim3 gg(TOKENS / 64, DIM / 64), gb(32, 4);
  k_qkv_gemm<<<gg, gb, 0, stream>>>(xbf, wqT, bq, qbuf, 0);
  k_qkv_gemm<<<gg, gb, 0, stream>>>(xbf, wkT, bk, kbuf, 0);
  k_qkv_gemm<<<gg, gb, 0, stream>>>(xbf, wvT, bv, vTbf, 1);

  k_attention<<<dim3(N_SEQ / 128, HEADS, B_SZ), dim3(32, 8), 0, stream>>>(
      qbuf, kbuf, vTbf, mask, aout);

  k_out_proj<<<gg, gb, 0, stream>>>(aout, woT, bo, out);
}
